// TransformerDecoderBlock_14568529068627
// MI455X (gfx1250) — matrix operand order repaired and checked by reading, not compiled, not run on hardware
//
#include <hip/hip_runtime.h>

typedef __bf16 bf16;
typedef __attribute__((ext_vector_type(16))) __bf16 v16bf;
typedef __attribute__((ext_vector_type(8)))  __bf16 v8bf;
typedef __attribute__((ext_vector_type(8)))  float  v8f;
typedef __attribute__((ext_vector_type(4)))  unsigned int u32x4;
typedef __attribute__((ext_vector_type(8)))  int i32x8;
typedef __attribute__((ext_vector_type(4)))  int i32x4;

#define WMMA_BF16(a, b, c) \
  __builtin_amdgcn_wmma_f32_16x16x32_bf16(false, (a), false, (b), (short)0, (c), false, false)

static constexpr int B_ = 2, S_ = 2048, D_ = 2048, HQ_ = 16, HKV_ = 4, DH_ = 128, P_ = 8192;

// ---------------------------------------------------------------------------
// Tensor Data Mover: async 2D bf16 tile load Global -> LDS.
// Tile: tile_rows rows of 32 bf16 elements (64B), row stride = row_elems*2 B.
// LDS dest: contiguous [row][32] (matches our fragment layout).
// Issue from ONE wave only; tracked by TENSORcnt.
// ---------------------------------------------------------------------------
__device__ __forceinline__ void tdm_load_tile(unsigned lds_off,
                                              const void* gaddr,
                                              unsigned tile_rows,
                                              unsigned row_elems) {
  unsigned long long ga = (unsigned long long)(uintptr_t)gaddr;
  u32x4 g0;
  g0[0] = 1u;                                           // count=1, user mode
  g0[1] = lds_off;                                      // LDS byte address
  g0[2] = (unsigned)(ga & 0xffffffffu);                 // global_addr[31:0]
  g0[3] = (unsigned)((ga >> 32) & 0x01ffffffu) | (2u << 30);  // [56:32], type=2

  const unsigned d0 = 0x40000000u;                      // tensor_dim0 (huge, no clip)
  const unsigned d1 = 0x40000000u;                      // tensor_dim1 (huge, no clip)
  const unsigned st = row_elems;                        // tensor_dim0_stride (elems)
  i32x8 g1;
  g1[0] = (int)(1u << 16);                              // data_size=1 (2 bytes)
  g1[1] = (int)((d0 & 0xffffu) << 16);                  // dim0[15:0] @ [63:48]
  g1[2] = (int)(((d0 >> 16) & 0xffffu) | ((d1 & 0xffffu) << 16));
  g1[3] = (int)(((d1 >> 16) & 0xffffu) | (32u << 16));  // tile_dim0 = 32 elems
  g1[4] = (int)(tile_rows & 0xffffu);                   // tile_dim1; tile_dim2=0
  g1[5] = (int)st;                                      // dim0_stride[31:0]
  g1[6] = (int)((unsigned long long)st >> 32);          // dim0_stride[47:32]; d1s lo=0
  g1[7] = 0;

  i32x4 z4 = {0, 0, 0, 0};
#if defined(__clang_major__) && (__clang_major__ >= 23)
  i32x8 z8 = {0, 0, 0, 0, 0, 0, 0, 0};
  __builtin_amdgcn_tensor_load_to_lds(g0, g1, z4, z4, z8, 0);
#else
  __builtin_amdgcn_tensor_load_to_lds(g0, g1, z4, z4, 0);
#endif
}

// ---------------------------------------------------------------------------
// RMSNorm: fp32 in -> bf16 out (one row per block)
// ---------------------------------------------------------------------------
__global__ __launch_bounds__(256) void rmsnorm_bf16_kernel(
    const float* __restrict__ x, const float* __restrict__ g,
    bf16* __restrict__ out, int ncols) {
  __shared__ float red[256];
  const int row = blockIdx.x;
  const float* xr = x + (size_t)row * ncols;
  float ss = 0.f;
  for (int i = threadIdx.x; i < ncols; i += 256) { float v = xr[i]; ss += v * v; }
  red[threadIdx.x] = ss;
  __syncthreads();
  for (int o = 128; o > 0; o >>= 1) {
    if (threadIdx.x < o) red[threadIdx.x] += red[threadIdx.x + o];
    __syncthreads();
  }
  const float scale = rsqrtf(red[0] / (float)ncols + 1e-5f);
  bf16* orow = out + (size_t)row * ncols;
  for (int i = threadIdx.x; i < ncols; i += 256)
    orow[i] = (bf16)(xr[i] * scale * g[i]);
}

// ---------------------------------------------------------------------------
// RoPE (rotate-half) fp32 -> bf16. Layout [b][s][h][128].
// ---------------------------------------------------------------------------
__global__ __launch_bounds__(256) void rope_bf16_kernel(
    const float* __restrict__ in, bf16* __restrict__ out, int H) {
  int idx = blockIdx.x * blockDim.x + threadIdx.x;
  int i = idx & 63; int t = idx >> 6;
  int h = t % H; t /= H;
  int s = t % S_; int b = t / S_;
  size_t base = (((size_t)b * S_ + s) * H + h) * DH_;
  float inv = __expf(-(float)i * (9.210340371976184f / 64.0f));  // ln(1e4)
  float ang = (float)s * inv;
  float sn, cs;
  __sincosf(ang, &sn, &cs);
  float v1 = in[base + i], v2 = in[base + i + 64];
  out[base + i]      = (bf16)(v1 * cs - v2 * sn);
  out[base + i + 64] = (bf16)(v1 * sn + v2 * cs);
}

// ---------------------------------------------------------------------------
// fp32 -> bf16 elementwise
// ---------------------------------------------------------------------------
__global__ __launch_bounds__(256) void f32_to_bf16_kernel(
    const float* __restrict__ in, bf16* __restrict__ out) {
  int i = blockIdx.x * blockDim.x + threadIdx.x;
  out[i] = (bf16)in[i];
}

// ---------------------------------------------------------------------------
// Weight pre-pass: fp32 [K x N] row-major -> bf16 transposed [N x K] row-major.
// LDS-tiled 32x32 so both global sides stay coalesced.
// ---------------------------------------------------------------------------
__global__ __launch_bounds__(256) void transpose_w_bf16_kernel(
    const float* __restrict__ W, bf16* __restrict__ Wt, int K, int N) {
  __shared__ bf16 t[32][33];
  const int kt = blockIdx.y * 32, nt = blockIdx.x * 32;
  const int lx = threadIdx.x & 31, ly = threadIdx.x >> 5;  // 32 x 8
#pragma unroll
  for (int yy = 0; yy < 32; yy += 8)
    t[yy + ly][lx] = (bf16)W[(size_t)(kt + yy + ly) * N + nt + lx];
  __syncthreads();
#pragma unroll
  for (int yy = 0; yy < 32; yy += 8)
    Wt[(size_t)(nt + yy + ly) * K + kt + lx] = t[lx][yy + ly];
}

// ---------------------------------------------------------------------------
// Tiled WMMA GEMM with TDM double-buffered tile streaming:
//   C[M x N] = A[M x K](bf16, row-major) * Wt[N x K](bf16, row-major)^T
//   workgroup tile 128x128, 8 waves, wave tile 32x64 (2x4 of 16x16)
// STORE_MODE 0: fp32 store   1: bf16 transposed V store (n->(g,dh), m->(b,s))
// ---------------------------------------------------------------------------
template <int STORE_MODE, bool ADD_BIAS, bool ADD_RES>
__global__ __launch_bounds__(256) void gemm_bf16_kernel(
    const bf16* __restrict__ A, const bf16* __restrict__ Wt,
    void* __restrict__ outp, const float* __restrict__ bias,
    const float* __restrict__ res, int K, int N) {
  __shared__ __align__(64) bf16 lds_a[2][128 * 32];  // [m][k]
  __shared__ __align__(64) bf16 lds_w[2][128 * 32];  // [n][k]

  const int tid = threadIdx.x;
  const int lane = tid & 31, wave = tid >> 5;
  const int lane16 = lane & 15, hl = lane >> 4;
  const int wm = wave >> 1, wn = wave & 1;
  const int M0 = blockIdx.y * 128, N0 = blockIdx.x * 128;

  v8f acc[2][4];
#pragma unroll
  for (int i = 0; i < 2; i++)
#pragma unroll
    for (int j = 0; j < 4; j++)
#pragma unroll
      for (int r = 0; r < 8; r++) acc[i][j][r] = 0.f;

  const int nsteps = K >> 5;
  if (tid < 32) {  // wave 0 drives the Tensor Data Mover
    tdm_load_tile((unsigned)(uintptr_t)&lds_a[0][0], A + (size_t)M0 * K, 128, (unsigned)K);
    tdm_load_tile((unsigned)(uintptr_t)&lds_w[0][0], Wt + (size_t)N0 * K, 128, (unsigned)K);
  }

  for (int it = 0; it < nsteps; it++) {
    const int buf = it & 1;
    if (tid < 32) {
      __builtin_amdgcn_s_wait_tensorcnt(0);  // current buffer landed
      if (it + 1 < nsteps) {                 // stream next tiles into other buffer
        const int k0 = (it + 1) << 5;
        tdm_load_tile((unsigned)(uintptr_t)&lds_a[buf ^ 1][0],
                      A + (size_t)M0 * K + k0, 128, (unsigned)K);
        tdm_load_tile((unsigned)(uintptr_t)&lds_w[buf ^ 1][0],
                      Wt + (size_t)N0 * K + k0, 128, (unsigned)K);
      }
    }
    __syncthreads();

    v16bf afr[2];
#pragma unroll
    for (int mt = 0; mt < 2; mt++) {
      const int m = wm * 32 + mt * 16 + lane16;
      const v8bf lo = *reinterpret_cast<const v8bf*>(&lds_a[buf][m * 32 + hl * 16]);
      const v8bf hi = *reinterpret_cast<const v8bf*>(&lds_a[buf][m * 32 + hl * 16 + 8]);
#pragma unroll
      for (int e = 0; e < 8; e++) { afr[mt][e] = lo[e]; afr[mt][8 + e] = hi[e]; }
    }
#pragma unroll
    for (int nt = 0; nt < 4; nt++) {
      const int n = wn * 64 + nt * 16 + lane16;
      const v16bf bfr = *reinterpret_cast<const v16bf*>(&lds_w[buf][n * 32 + hl * 16]);
#pragma unroll
      for (int mt = 0; mt < 2; mt++)
        acc[mt][nt] = WMMA_BF16(afr[mt], bfr, acc[mt][nt]);
    }
    __syncthreads();  // compute(it) done before TDM overwrites this buffer
  }

  // epilogue
#pragma unroll
  for (int mt = 0; mt < 2; mt++) {
#pragma unroll
    for (int nt = 0; nt < 4; nt++) {
      const int n = N0 + wn * 64 + nt * 16 + lane16;
      float bs = 0.f;
      if constexpr (ADD_BIAS) bs = bias[n];
#pragma unroll
      for (int r = 0; r < 8; r++) {
        const int m = M0 + wm * 32 + mt * 16 + r + 8 * hl;
        float v = acc[mt][nt][r] + bs;
        if constexpr (ADD_RES) v += res[(size_t)m * N + n];
        if constexpr (STORE_MODE == 0) {
          ((float*)outp)[(size_t)m * N + n] = v;
        } else {  // transposed bf16 store for V: out[(b,g,dh)][s]
          const int b = m >> 11, s = m & 2047;
          const int g = n >> 7, dh = n & 127;
          ((bf16*)outp)[(((size_t)b * HKV_ + g) * DH_ + dh) * S_ + s] = (bf16)v;
        }
      }
    }
  }
}

// ---------------------------------------------------------------------------
// Fused SwiGLU dual GEMM with TDM streaming:
//   act = silu(A*Wg + bg) * (A*Wu + bu), bf16 out.
//   workgroup tile 64(M) x 128(N), 8 waves (2x4), wave tile 32x32.
//   WgT/WuT are bf16 [P x D] row-major (pre-transposed).
// ---------------------------------------------------------------------------
__global__ __launch_bounds__(256) void glu_dual_kernel(
    const bf16* __restrict__ A, const bf16* __restrict__ WgT,
    const bf16* __restrict__ WuT, const float* __restrict__ bg,
    const float* __restrict__ bu, bf16* __restrict__ act) {
  __shared__ __align__(64) bf16 lds_a[2][64 * 32];
  __shared__ __align__(64) bf16 lds_g[2][128 * 32];
  __shared__ __align__(64) bf16 lds_u[2][128 * 32];

  const int tid = threadIdx.x;
  const int lane = tid & 31, wave = tid >> 5;
  const int lane16 = lane & 15, hl = lane >> 4;
  const int wm = wave >> 2, wn = wave & 3;
  const int M0 = blockIdx.y * 64, N0 = blockIdx.x * 128;

  v8f ag[2][2], au[2][2];
#pragma unroll
  for (int i = 0; i < 2; i++)
#pragma unroll
    for (int j = 0; j < 2; j++)
#pragma unroll
      for (int r = 0; r < 8; r++) { ag[i][j][r] = 0.f; au[i][j][r] = 0.f; }

  const int nsteps = D_ >> 5;
  if (tid < 32) {
    tdm_load_tile((unsigned)(uintptr_t)&lds_a[0][0], A + (size_t)M0 * D_, 64, D_);
    tdm_load_tile((unsigned)(uintptr_t)&lds_g[0][0], WgT + (size_t)N0 * D_, 128, D_);
    tdm_load_tile((unsigned)(uintptr_t)&lds_u[0][0], WuT + (size_t)N0 * D_, 128, D_);
  }

  for (int it = 0; it < nsteps; it++) {
    const int buf = it & 1;
    if (tid < 32) {
      __builtin_amdgcn_s_wait_tensorcnt(0);
      if (it + 1 < nsteps) {
        const int k0 = (it + 1) << 5;
        tdm_load_tile((unsigned)(uintptr_t)&lds_a[buf ^ 1][0],
                      A + (size_t)M0 * D_ + k0, 64, D_);
        tdm_load_tile((unsigned)(uintptr_t)&lds_g[buf ^ 1][0],
                      WgT + (size_t)N0 * D_ + k0, 128, D_);
        tdm_load_tile((unsigned)(uintptr_t)&lds_u[buf ^ 1][0],
                      WuT + (size_t)N0 * D_ + k0, 128, D_);
      }
    }
    __syncthreads();

    v16bf afr[2];
#pragma unroll
    for (int mt = 0; mt < 2; mt++) {
      const int m = wm * 32 + mt * 16 + lane16;
      const v8bf lo = *reinterpret_cast<const v8bf*>(&lds_a[buf][m * 32 + hl * 16]);
      const v8bf hi = *reinterpret_cast<const v8bf*>(&lds_a[buf][m * 32 + hl * 16 + 8]);
#pragma unroll
      for (int e = 0; e < 8; e++) { afr[mt][e] = lo[e]; afr[mt][8 + e] = hi[e]; }
    }
#pragma unroll
    for (int nt = 0; nt < 2; nt++) {
      const int n = wn * 32 + nt * 16 + lane16;
      const v16bf bg_fr = *reinterpret_cast<const v16bf*>(&lds_g[buf][n * 32 + hl * 16]);
      const v16bf bu_fr = *reinterpret_cast<const v16bf*>(&lds_u[buf][n * 32 + hl * 16]);
#pragma unroll
      for (int mt = 0; mt < 2; mt++) {
        ag[mt][nt] = WMMA_BF16(afr[mt], bg_fr, ag[mt][nt]);
        au[mt][nt] = WMMA_BF16(afr[mt], bu_fr, au[mt][nt]);
      }
    }
    __syncthreads();
  }

#pragma unroll
  for (int mt = 0; mt < 2; mt++) {
#pragma unroll
    for (int nt = 0; nt < 2; nt++) {
      const int n = N0 + wn * 32 + nt * 16 + lane16;
      const float bgv = bg[n], buv = bu[n];
#pragma unroll
      for (int r = 0; r < 8; r++) {
        const int m = M0 + wm * 32 + mt * 16 + r + 8 * hl;
        const float g = ag[mt][nt][r] + bgv;
        const float u = au[mt][nt][r] + buv;
        const float sv = g / (1.f + __expf(-g)) * u;  // silu(g)*u
        act[(size_t)m * P_ + n] = (bf16)sv;
      }
    }
  }
}

// ---------------------------------------------------------------------------
// Flash attention (causal GQA). One wave per (b, head, 16-query tile).
// Q: [b][s][hq][128] bf16 (RoPE'd), K: [b][s][g][128] bf16 (RoPE'd),
// Vt: [b][g][dh][s] bf16, O: [b][s][hq][128] fp32.
// ---------------------------------------------------------------------------
__global__ __launch_bounds__(32) void flash_attn_kernel(
    const bf16* __restrict__ Q, const bf16* __restrict__ Kc,
    const bf16* __restrict__ Vt, float* __restrict__ O) {
  __shared__ __align__(32) bf16 p_lds[16 * 32];

  const int lane = threadIdx.x;
  const int lane16 = lane & 15, hl = lane >> 4;
  const int q0 = blockIdx.x * 16;
  const int bh = blockIdx.y;
  const int b = bh / HQ_, h = bh % HQ_;
  const int g = h % HKV_;
  const float scale = 0.08838834764831845f;  // 1/sqrt(128)

  v16bf qf[4];
  {
    const bf16* qbase = Q + (((size_t)b * S_ + q0 + lane16) * D_ + h * DH_);
#pragma unroll
    for (int c = 0; c < 4; c++) {
      const v8bf lo = *reinterpret_cast<const v8bf*>(qbase + c * 32 + hl * 16);
      const v8bf hi = *reinterpret_cast<const v8bf*>(qbase + c * 32 + hl * 16 + 8);
#pragma unroll
      for (int e = 0; e < 8; e++) { qf[c][e] = lo[e]; qf[c][8 + e] = hi[e]; }
    }
  }

  v8f out[8];
#pragma unroll
  for (int nt = 0; nt < 8; nt++)
#pragma unroll
    for (int r = 0; r < 8; r++) out[nt][r] = 0.f;
  float mrow[8], lrow[8];
#pragma unroll
  for (int r = 0; r < 8; r++) { mrow[r] = -3.0e38f; lrow[r] = 0.f; }

  const bf16* kbase = Kc + (size_t)b * S_ * (HKV_ * DH_) + g * DH_;
  const bf16* vbase = Vt + (((size_t)b * HKV_ + g) * DH_) * S_;

  const int nblk = (q0 + 47) >> 5;
  for (int kb = 0; kb < nblk; kb++) {
    const int kk = kb * 32;
    if (kb + 1 < nblk)  // prefetch next key block (global_prefetch_b8)
      __builtin_prefetch(kbase + (size_t)(kk + 32 + lane16) * (HKV_ * DH_), 0, 3);
    v8f s0, s1;
#pragma unroll
    for (int r = 0; r < 8; r++) { s0[r] = 0.f; s1[r] = 0.f; }
#pragma unroll
    for (int c = 0; c < 4; c++) {
      const v16bf k0f = *reinterpret_cast<const v16bf*>(
          kbase + (size_t)(kk + lane16) * (HKV_ * DH_) + c * 32 + hl * 16);
      const v16bf k1f = *reinterpret_cast<const v16bf*>(
          kbase + (size_t)(kk + 16 + lane16) * (HKV_ * DH_) + c * 32 + hl * 16);
      s0 = WMMA_BF16(qf[c], k0f, s0);
      s1 = WMMA_BF16(qf[c], k1f, s1);
    }
    const bool need_mask = (kk + 31 > q0);
    float alpha[8];
#pragma unroll
    for (int r = 0; r < 8; r++) {
      float a = s0[r] * scale, bb = s1[r] * scale;
      if (need_mask) {
        const int q = q0 + r + 8 * hl;
        if (kk + lane16 > q) a = -1.0e30f;
        if (kk + 16 + lane16 > q) bb = -1.0e30f;
      }
      float t = fmaxf(a, bb);
#pragma unroll
      for (int msk = 8; msk > 0; msk >>= 1) t = fmaxf(t, __shfl_xor(t, msk, 16));
      const float mnew = fmaxf(mrow[r], t);
      alpha[r] = __expf(mrow[r] - mnew);
      mrow[r] = mnew;
      const float p0 = __expf(a - mnew), p1 = __expf(bb - mnew);
      s0[r] = p0; s1[r] = p1;
      float rs = p0 + p1;
#pragma unroll
      for (int msk = 8; msk > 0; msk >>= 1) rs += __shfl_xor(rs, msk, 16);
      lrow[r] = lrow[r] * alpha[r] + rs;
    }
    // transpose P (C-layout -> A-layout) through LDS
#pragma unroll
    for (int r = 0; r < 8; r++) {
      p_lds[(r + 8 * hl) * 32 + lane16] = (bf16)s0[r];
      p_lds[(r + 8 * hl) * 32 + 16 + lane16] = (bf16)s1[r];
    }
    __syncthreads();
    v16bf pf;
    {
      const v8bf lo = *reinterpret_cast<const v8bf*>(&p_lds[lane16 * 32 + hl * 16]);
      const v8bf hi = *reinterpret_cast<const v8bf*>(&p_lds[lane16 * 32 + hl * 16 + 8]);
#pragma unroll
      for (int e = 0; e < 8; e++) { pf[e] = lo[e]; pf[8 + e] = hi[e]; }
    }
    __syncthreads();
#pragma unroll
    for (int nt = 0; nt < 8; nt++) {
      const v16bf vf = *reinterpret_cast<const v16bf*>(
          vbase + (size_t)(nt * 16 + lane16) * S_ + kk + hl * 16);
#pragma unroll
      for (int r = 0; r < 8; r++) out[nt][r] *= alpha[r];
      out[nt] = WMMA_BF16(pf, vf, out[nt]);
    }
  }

  float* obase = O + ((size_t)b * S_) * D_ + h * DH_;
#pragma unroll
  for (int nt = 0; nt < 8; nt++) {
#pragma unroll
    for (int r = 0; r < 8; r++) {
      const int q = q0 + r + 8 * hl;
      obase[(size_t)q * D_ + nt * 16 + lane16] = out[nt][r] / lrow[r];
    }
  }
}

// ---------------------------------------------------------------------------
extern "C" void kernel_launch(void* const* d_in, const int* in_sizes, int n_in,
                              void* d_out, int out_size, void* d_ws, size_t ws_size,
                              hipStream_t stream) {
  const float* x  = (const float*)d_in[0];
  // d_in[1] = attention_mask (causal; applied analytically in-kernel)
  const float* Wq = (const float*)d_in[2];
  const float* Wk = (const float*)d_in[3];
  const float* Wv = (const float*)d_in[4];
  const float* Wo = (const float*)d_in[5];
  const float* Wg = (const float*)d_in[6];
  const float* bg = (const float*)d_in[7];
  const float* Wu = (const float*)d_in[8];
  const float* bu = (const float*)d_in[9];
  const float* Wd = (const float*)d_in[10];
  const float* bd = (const float*)d_in[11];
  const float* g1 = (const float*)d_in[12];
  const float* g2 = (const float*)d_in[13];

  const size_t M = (size_t)B_ * S_;  // 4096 rows

  char* p = (char*)d_ws;
  auto carve = [&](size_t bytes) -> void* {
    void* r = (void*)p;
    p += (bytes + 255) & ~(size_t)255;
    return r;
  };
  bf16*  xn  = (bf16*)carve(M * D_ * sizeof(bf16));            // xn / hn (reused)
  float* tmp = (float*)carve(M * D_ * sizeof(float));          // Qf32 / Kf32 / o_f32
  bf16*  qb  = (bf16*)carve(M * D_ * sizeof(bf16));            // Q bf16 / o bf16
  bf16*  kb  = (bf16*)carve(M * (HKV_ * DH_) * sizeof(bf16));  // K bf16
  bf16*  vt  = (bf16*)carve(M * (HKV_ * DH_) * sizeof(bf16));  // V^T bf16
  float* hbuf= (float*)carve(M * D_ * sizeof(float));          // h = x + o@Wo
  bf16*  act = (bf16*)carve(M * (size_t)P_ * sizeof(bf16));    // silu(g)*u bf16
  // pre-transposed bf16 weights [N x K]
  bf16* wqT = (bf16*)carve((size_t)D_ * D_ * sizeof(bf16));
  bf16* wkT = (bf16*)carve((size_t)(HKV_ * DH_) * D_ * sizeof(bf16));
  bf16* wvT = (bf16*)carve((size_t)(HKV_ * DH_) * D_ * sizeof(bf16));
  bf16* woT = (bf16*)carve((size_t)D_ * D_ * sizeof(bf16));
  bf16* wgT = (bf16*)carve((size_t)P_ * D_ * sizeof(bf16));
  bf16* wuT = (bf16*)carve((size_t)P_ * D_ * sizeof(bf16));
  bf16* wdT = (bf16*)carve((size_t)D_ * P_ * sizeof(bf16));

  // 0) weight pre-pass: fp32 [K x N] -> bf16 [N x K]
  transpose_w_bf16_kernel<<<dim3(D_ / 32, D_ / 32), 256, 0, stream>>>(Wq, wqT, D_, D_);
  transpose_w_bf16_kernel<<<dim3((HKV_ * DH_) / 32, D_ / 32), 256, 0, stream>>>(Wk, wkT, D_, HKV_ * DH_);
  transpose_w_bf16_kernel<<<dim3((HKV_ * DH_) / 32, D_ / 32), 256, 0, stream>>>(Wv, wvT, D_, HKV_ * DH_);
  transpose_w_bf16_kernel<<<dim3(D_ / 32, D_ / 32), 256, 0, stream>>>(Wo, woT, D_, D_);
  transpose_w_bf16_kernel<<<dim3(P_ / 32, D_ / 32), 256, 0, stream>>>(Wg, wgT, D_, P_);
  transpose_w_bf16_kernel<<<dim3(P_ / 32, D_ / 32), 256, 0, stream>>>(Wu, wuT, D_, P_);
  transpose_w_bf16_kernel<<<dim3(D_ / 32, P_ / 32), 256, 0, stream>>>(Wd, wdT, P_, D_);

  // 1) xn = rmsnorm(x, g1)
  rmsnorm_bf16_kernel<<<(int)M, 256, 0, stream>>>(x, g1, xn, D_);
  // 2) Q = xn @ Wq  (fp32)
  gemm_bf16_kernel<0, false, false><<<dim3(D_ / 128, M / 128), 256, 0, stream>>>(
      xn, wqT, tmp, nullptr, nullptr, D_, D_);
  // 3) RoPE(Q) -> bf16
  rope_bf16_kernel<<<(int)(M * HQ_ * 64 / 256), 256, 0, stream>>>(tmp, qb, HQ_);
  // 4) K = xn @ Wk (fp32)
  gemm_bf16_kernel<0, false, false><<<dim3((HKV_ * DH_) / 128, M / 128), 256, 0, stream>>>(
      xn, wkT, tmp, nullptr, nullptr, D_, HKV_ * DH_);
  // 5) RoPE(K) -> bf16
  rope_bf16_kernel<<<(int)(M * HKV_ * 64 / 256), 256, 0, stream>>>(tmp, kb, HKV_);
  // 6) V^T = (xn @ Wv)^T bf16, stored transposed
  gemm_bf16_kernel<1, false, false><<<dim3((HKV_ * DH_) / 128, M / 128), 256, 0, stream>>>(
      xn, wvT, vt, nullptr, nullptr, D_, HKV_ * DH_);
  // 7) causal flash attention -> o fp32 in tmp
  flash_attn_kernel<<<dim3(S_ / 16, B_ * HQ_), 32, 0, stream>>>(qb, kb, vt, tmp);
  // 8) o -> bf16
  f32_to_bf16_kernel<<<(int)(M * D_ / 256), 256, 0, stream>>>(tmp, qb);
  // 9) h = x + o @ Wo
  gemm_bf16_kernel<0, false, true><<<dim3(D_ / 128, M / 128), 256, 0, stream>>>(
      qb, woT, hbuf, nullptr, x, D_, D_);
  // 10) hn = rmsnorm(h, g2)
  rmsnorm_bf16_kernel<<<(int)M, 256, 0, stream>>>(hbuf, g2, xn, D_);
  // 11) act = silu(hn@Wg+bg) * (hn@Wu+bu)  (fused dual GEMM)
  glu_dual_kernel<<<dim3(P_ / 128, M / 64), 256, 0, stream>>>(xn, wgT, wuT, bg, bu, act);
  // 12) out = h + act @ Wd + bd
  gemm_bf16_kernel<0, true, true><<<dim3(D_ / 128, M / 128), 256, 0, stream>>>(
      act, wdT, (float*)d_out, bd, hbuf, P_, D_);
}
